// GumbelVectorQuantizer_57011395887453
// MI455X (gfx1250) — matrix operand order repaired and checked by reading, not compiled, not run on hardware
//
#include <hip/hip_runtime.h>
#include <hip/hip_bf16.h>

typedef __attribute__((ext_vector_type(16))) _Float16 v16h;
typedef __attribute__((ext_vector_type(8)))  _Float16 v8h;
typedef __attribute__((ext_vector_type(8)))  float    v8f;

#define GVQ_EPS 1e-10f

// Problem dims (fixed by reference)
constexpr int Din   = 512;
constexpr int Gn    = 2;
constexpr int Cn    = 320;
constexpr int Dg    = 128;          // D_out / G
constexpr int Ncol  = Gn * Cn;      // 640
constexpr int Mtot  = 16 * 4096;    // B * S = 65536 rows
constexpr int Bb    = 16;           // batch
// Tiling
constexpr int BM    = 32;           // rows per block (two WMMA M subtiles)
constexpr int MSUB  = 2;            // M subtiles sharing each B fragment
constexpr int WAVES = 8;            // 256 threads
constexpr int NPW   = Ncol / WAVES; // 80 cols per wave
constexpr int NT    = NPW / 16;     // 5 WMMA N tiles per wave

// LDS layout (static, ~99.3 KB + usage partials; CU-mode LDS limit is 160 KB)
constexpr int AROW_BYTES  = Din * 4 + 16;      // 2064: padded f32 A row (bank spread)
constexpr int ASLAB_BYTES = BM * AROW_BYTES;   // 66048 (dies after f16 conversion)
constexpr int LGT_BYTES   = BM * Ncol * 2;     // 40960 f16 logits (alias f32 slab)
constexpr int AH_OFF      = ASLAB_BYTES;       // f16 A slab after the f32 slab
constexpr int AH_ROW      = Din * 2 + 16;      // 1040: padded f16 A row
constexpr int SH_BYTES    = AH_OFF + BM * AH_ROW;   // 99328

static_assert(LGT_BYTES <= ASLAB_BYTES, "logits must fit in dead f32-slab region");

__global__ __launch_bounds__(256)
void gvq_main(const float* __restrict__ X,          // [65536, 512]
              const long long* __restrict__ vlen,   // [16] int64
              const float* __restrict__ U,          // [65536, 640] noise
              const _Float16* __restrict__ Wh,      // [640, 512] pre-converted f16
              const float* __restrict__ bias,       // [640]
              const float* __restrict__ CB,         // [2, 320, 128]
              const float* __restrict__ Tptr,       // [1]
              float* __restrict__ outQ,             // [65536, 256]
              float* __restrict__ usage_acc)        // [640] global partials
{
    __shared__ __align__(16) char shbuf[SH_BYTES];
    __shared__ float us[Ncol];                     // block-local usage partials

    const int tid  = threadIdx.x;
    const int lane = tid & 31;
    const int w    = tid >> 5;
    const int half = lane >> 4;            // 0: lanes 0-15, 1: lanes 16-31
    const int l15  = lane & 15;
    const int r0   = blockIdx.x * BM;      // global row base

    // ---- Async DMA: whole 32x512 f32 A slab -> LDS (ASYNCcnt-tracked) --------
    {
        const unsigned lds_base = (unsigned)(size_t)(void*)shbuf;
        #pragma unroll
        for (int c = 0; c < 16; ++c) {
            const int f      = c * 256 + tid;       // 16-byte chunk id, 0..4095
            const int row    = f >> 7;              // 128 chunks per row
            const int inrow  = (f & 127) << 4;      // byte offset within row
            const unsigned ldsa = lds_base + row * AROW_BYTES + inrow;
            const float* gp = X + (size_t)(r0 + row) * Din + (inrow >> 2);
            asm volatile("global_load_async_to_lds_b128 %0, %1, off"
                         :: "v"(ldsa), "v"(gp)
                         : "memory");
        }
    }

    for (int i = tid; i < Ncol; i += 256) us[i] = 0.0f;   // overlap with DMA

    v8f acc[MSUB][NT];
    #pragma unroll
    for (int m = 0; m < MSUB; ++m)
        #pragma unroll
        for (int t = 0; t < NT; ++t)
            #pragma unroll
            for (int j = 0; j < 8; ++j) acc[m][t][j] = 0.0f;

    asm volatile("s_wait_asynccnt 0x0" ::: "memory");
    __syncthreads();

    // ---- One-shot LDS f32 -> f16 conversion of the A slab --------------------
    for (int q = tid; q < BM * (Din / 8); q += 256) {     // 2048 chunks of 8 elems
        const int row = q >> 6;                            // 64 chunks per row
        const int k   = (q & 63) * 8;
        const float*   src = (const float*)(shbuf + row * AROW_BYTES + k * 4);
        _Float16*      dst = (_Float16*)(shbuf + AH_OFF + row * AH_ROW + k * 2);
        #pragma unroll
        for (int j = 0; j < 8; ++j) dst[j] = (_Float16)src[j];
    }
    __syncthreads();

    // ---------------- Phase 1: GEMM logits = X @ W^T + b (f16 WMMA, f32 acc) ----
    // Two M subtiles per block share every B fragment (halves W traffic).
    const char* aH0 = shbuf + AH_OFF + l15 * AH_ROW;           // rows 0..15
    const char* aH1 = aH0 + 16 * AH_ROW;                       // rows 16..31
    for (int k0 = 0; k0 < Din; k0 += 32) {
        // A fragment (16x32 f16): lane row = l15.
        // lanes 0-15 hold K 0..15; lanes 16-31 hold K 16..31 (contiguous)
        const size_t o0 = (size_t)(k0 + half * 16) * 2;
        const size_t o1 = (size_t)(k0 + half * 16 + 8) * 2;
        const v8h lo0 = *(const v8h*)(aH0 + o0);
        const v8h hi0 = *(const v8h*)(aH0 + o1);
        const v8h lo1 = *(const v8h*)(aH1 + o0);
        const v8h hi1 = *(const v8h*)(aH1 + o1);
        const v16h af0 = __builtin_shufflevector(lo0, hi0,
            0, 1, 2, 3, 4, 5, 6, 7, 8, 9, 10, 11, 12, 13, 14, 15);
        const v16h af1 = __builtin_shufflevector(lo1, hi1,
            0, 1, 2, 3, 4, 5, 6, 7, 8, 9, 10, 11, 12, 13, 14, 15);

        #pragma unroll
        for (int t = 0; t < NT; ++t) {
            // B fragment (32x16 f16): lane col = l15 (output channel of W);
            // lanes 0-15 hold K 0..15, lanes 16-31 hold K 16..31 (contiguous!)
            const int col = w * NPW + t * 16 + l15;
            const v16h bfrag = *(const v16h*)(Wh + (size_t)col * Din + k0 + half * 16);
            acc[0][t] = __builtin_amdgcn_wmma_f32_16x16x32_f16(
                false, af0, false, bfrag, (short)0, acc[0][t], false, false);
            acc[1][t] = __builtin_amdgcn_wmma_f32_16x16x32_f16(
                false, af1, false, bfrag, (short)0, acc[1][t], false, false);
        }
    }
    __syncthreads();   // A slabs dead; f16 logits buffer takes over the front region

    // Epilogue: C/D layout -> LDS. VGPR j, lane: M = j + 8*half, N = l15.
    _Float16 (*lgt)[Ncol] = (_Float16 (*)[Ncol])shbuf;
    #pragma unroll
    for (int m = 0; m < MSUB; ++m)
        #pragma unroll
        for (int t = 0; t < NT; ++t) {
            const int col = w * NPW + t * 16 + l15;
            const float bv = bias[col];
            #pragma unroll
            for (int j = 0; j < 8; ++j)
                lgt[m * 16 + j + 8 * half][col] = (_Float16)(acc[m][t][j] + bv);
        }
    __syncthreads();

    // ---------------- Phase 2: gumbel-softmax + argmax + usage + codebook ------
    const float invT = 1.0f / Tptr[0];
    for (int task = w; task < BM * Gn; task += WAVES) {
        const int row  = task >> 1;
        const int g    = task & 1;
        const int grow = r0 + row;
        const int bi   = grow >> 12;        // / 4096
        const int sp   = grow & 4095;
        const bool valid = (long long)sp < vlen[bi];
        const float* urow = U + (size_t)grow * Ncol + g * Cn;

        // prefetch next task's noise row while we crunch this one
        const int task2 = task + WAVES;
        if (task2 < BM * Gn) {
            const float* un = U + (size_t)(r0 + (task2 >> 1)) * Ncol + (task2 & 1) * Cn;
            __builtin_prefetch(un + lane * 10, 0, 0);
        }

        float lv[10], gv[10], sp2[10];
        float bm = -3.4e38f; int bidx = 0;  // argmax over (logit + gumbel)
        float m2 = -3.4e38f;                // max over plain logits
        #pragma unroll
        for (int i = 0; i < 10; ++i) {
            const int c = lane + 32 * i;
            const float l  = (float)lgt[row][g * Cn + c];
            const float u  = urow[c];
            const float gb = -__logf(-__logf(u + GVQ_EPS) + GVQ_EPS);
            lv[i] = l; gv[i] = l + gb;
            if (gv[i] > bm) { bm = gv[i]; bidx = c; }
            m2 = fmaxf(m2, l);
        }
        // wave32 reductions: argmax (first-index tiebreak), max, then sums
        #pragma unroll
        for (int off = 16; off > 0; off >>= 1) {
            const float ov = __shfl_xor(bm, off, 32);
            const int   oi = __shfl_xor(bidx, off, 32);
            if (ov > bm || (ov == bm && oi < bidx)) { bm = ov; bidx = oi; }
            m2 = fmaxf(m2, __shfl_xor(m2, off, 32));
        }
        float s1 = 0.0f, s2 = 0.0f;
        #pragma unroll
        for (int i = 0; i < 10; ++i) {
            s1 += __expf((gv[i] - bm) * invT);
            sp2[i] = __expf(lv[i] - m2);
            s2 += sp2[i];
        }
        #pragma unroll
        for (int off = 16; off > 0; off >>= 1) {
            s1 += __shfl_xor(s1, off, 32);
            s2 += __shfl_xor(s2, off, 32);
        }
        // masked usage partials (softmax of plain logits) into LDS
        if (valid) {
            const float inv_s2 = 1.0f / s2;
            #pragma unroll
            for (int i = 0; i < 10; ++i)
                atomicAdd(&us[g * Cn + (lane + 32 * i)], sp2[i] * inv_s2);
        }
        // straight-through: forward value of choice_probs is one-hot scaled by (1-y)+y
        const float y  = 1.0f / s1;         // y_soft at its own argmax
        const float cp = (1.0f - y) + y;
        const float* cb = CB + ((size_t)g * Cn + bidx) * Dg;
        float* op = outQ + (size_t)grow * (Gn * Dg) + g * Dg;
        #pragma unroll
        for (int d = 0; d < Dg; d += 32) op[d + lane] = cp * cb[d + lane];
    }
    __syncthreads();

    // flush block usage partials to global
    for (int i = tid; i < Ncol; i += 256) atomicAdd(&usage_acc[i], us[i]);
}

// One-shot W f32 -> f16 conversion (0.65 MB, lives in L2 across the main grid)
__global__ void gvq_cvtw(const float* __restrict__ W, _Float16* __restrict__ Wh) {
    const int i = blockIdx.x * blockDim.x + threadIdx.x;
    if (i < Ncol * Din) Wh[i] = (_Float16)W[i];
}

__global__ void gvq_zero(float* __restrict__ usage_acc) {
    const int i = blockIdx.x * blockDim.x + threadIdx.x;
    if (i < Ncol) usage_acc[i] = 0.0f;
}

__global__ void gvq_final(const float* __restrict__ usage_acc,
                          const long long* __restrict__ vlen,
                          float* __restrict__ outU) {
    const int i = blockIdx.x * blockDim.x + threadIdx.x;
    if (i < Ncol) {
        float tot = 0.0f;
        for (int b = 0; b < Bb; ++b) tot += (float)vlen[b];
        outU[i] = usage_acc[i] / tot;
    }
}

extern "C" void kernel_launch(void* const* d_in, const int* in_sizes, int n_in,
                              void* d_out, int out_size, void* d_ws, size_t ws_size,
                              hipStream_t stream) {
    const float*     X    = (const float*)d_in[0];
    const long long* vlen = (const long long*)d_in[1];   // int64 per reference
    const float*     U    = (const float*)d_in[2];
    const float*     W    = (const float*)d_in[3];
    const float*     bias = (const float*)d_in[4];
    const float*     CB   = (const float*)d_in[5];
    const float*     T    = (const float*)d_in[6];

    float* outQ      = (float*)d_out;                    // [65536, 256]
    float* outU      = outQ + (size_t)Mtot * (Gn * Dg);  // [640] appended
    float* usage_acc = (float*)d_ws;                     // 640 f32 at offset 0
    _Float16* Wh     = (_Float16*)((char*)d_ws + 4096);  // 640*512 f16 scratch

    gvq_cvtw<<<(Ncol * Din + 255) / 256, 256, 0, stream>>>(W, Wh);
    gvq_zero<<<(Ncol + 255) / 256, 256, 0, stream>>>(usage_acc);
    gvq_main<<<Mtot / BM, 256, 0, stream>>>(X, vlen, U, Wh, bias, CB, T, outQ, usage_acc);
    gvq_final<<<(Ncol + 255) / 256, 256, 0, stream>>>(usage_acc, vlen, outU);
}
